// PersonReIDModel_24962349924742
// MI455X (gfx1250) — matrix operand order repaired and checked by reading, not compiled, not run on hardware
//
#include <hip/hip_runtime.h>
#include <hip/hip_bf16.h>
#include <math.h>

typedef __attribute__((ext_vector_type(16))) _Float16 v16h;
typedef __attribute__((ext_vector_type(8)))  _Float16 v8h;
typedef __attribute__((ext_vector_type(8)))  float    v8f;

static constexpr int TBN = 128;   // N tile (b*oh*ow columns)
static constexpr int TBK = 32;    // K step (matches wmma f16 K)
// M tile = 64*MT (MT=1 for OC=64 layers, MT=2 for OC>=128 layers)

// ---------------------------------------------------------------------------
// Implicit-GEMM convolution with fused BN (scale/bias), residual add, ReLU.
// A = weights [OC][K] (f16, pre-converted), B = on-the-fly im2col (f32->f16),
// C accumulated in f32 via v_wmma_f32_16x16x32_f16.
// Input activation layout: idx = ic*sc + b*sb + h*W + w  (supports NCHW and CBHW)
// Output layout: out[oc][b][oh][ow]  (CBHW with columns n = b*OH*OW + oh*OW + ow)
// Wave layout: 8 wave32s as 4(M) x 2(N); each wave: MT M-tiles x 4 N-tiles,
// B fragments reused across the MT M-tiles in registers.
// ---------------------------------------------------------------------------
template <int MT>
__global__ __launch_bounds__(256) void conv_wmma_kernel(
    const float* __restrict__ in, int sb, int sc, int H, int W,
    const _Float16* __restrict__ wgt, int OC, int K,
    int KHKW, int KW, int stride, int pad,
    int OH, int OW, int Ntot,
    const float* __restrict__ scale, const float* __restrict__ bias,
    const float* __restrict__ res, int do_relu,
    float* __restrict__ out)
{
  constexpr int TBM = 64 * MT;
  __shared__ alignas(32) _Float16 lsA[TBM * TBK];   // [m][k] row-major
  __shared__ alignas(32) _Float16 lsB[TBN * TBK];   // [n][k] (B transposed)

  const int tid    = threadIdx.x;
  const int lane   = tid & 31;
  const int wid    = tid >> 5;        // 8 waves
  const int lh     = lane >> 4;       // lane half (0/1)
  const int l16    = lane & 15;
  const int wm     = wid >> 1;        // wave M-group 0..3 (owns MT tiles)
  const int wc     = wid & 1;         // wave N-half 0..1

  const int nBlock  = blockIdx.x * TBN;
  const int ocBlock = blockIdx.y * TBM;
  const int OHOW    = OH * OW;

  // global->LDS thread mappings
  const int arow = tid >> 2;          // 0..63  (A: weight row; +64 for MT=2)
  const int ak0  = (tid & 3) * 8;     // 8 f16 per thread per row
  const int kr   = tid >> 3;          // 0..31  (B: k row)
  const int nb0  = (tid & 7) * 16;    // 16 columns per thread

  v8f acc[MT][4] = {};

  const int kSteps = (K + TBK - 1) / TBK;
  for (int kb = 0; kb < kSteps; ++kb) {
    // ---- stage A: weights (already f16); OC is divisible by 64*MT ----
    {
      const int gk = kb * TBK + ak0;
#pragma unroll
      for (int s = 0; s < MT; ++s) {
        const int row = arow + s * 64;
        const _Float16* wp = wgt + (size_t)(ocBlock + row) * K + gk;
#pragma unroll
        for (int j = 0; j < 8; ++j)
          lsA[row * TBK + ak0 + j] = (gk + j < K) ? wp[j] : (_Float16)0.0f;
        if (kb + 1 < kSteps) __builtin_prefetch(wp + TBK, 0, 0);
      }
    }
    // ---- stage B: implicit im2col gather, f32 -> f16 ----
    {
      const int k = kb * TBK + kr;
      const bool vk = (k < K);
      int ic = 0, khh = 0, kww = 0;
      if (vk) { ic = k / KHKW; int kk = k - ic * KHKW; khh = kk / KW; kww = kk - khh * KW; }
      int n = nBlock + nb0;
      int b = n / OHOW; int rem = n - b * OHOW;
      int oh = rem / OW; int ow = rem - oh * OW;
      const float* inc_ = in + (size_t)ic * sc;
#pragma unroll
      for (int j = 0; j < 16; ++j) {
        float v = 0.0f;
        if (vk && (n + j) < Ntot) {
          const int h = oh * stride - pad + khh;
          const int w = ow * stride - pad + kww;
          if (h >= 0 && h < H && w >= 0 && w < W)
            v = inc_[b * sb + h * W + w];
        }
        lsB[(nb0 + j) * TBK + kr] = (_Float16)v;
        if (++ow == OW) { ow = 0; if (++oh == OH) { oh = 0; ++b; } }
      }
    }
    __syncthreads();
    // ---- compute: MT*4 WMMAs per wave; B fragment reused across M tiles ----
    {
      // A fragment 16x32 f16: lane half 0 -> K 0..15, half 1 -> 16..31
      v16h aFrag[MT];
#pragma unroll
      for (int s = 0; s < MT; ++s) {
        const _Float16* ap = &lsA[((wm * MT + s) * 16 + l16) * TBK + lh * 16];
        v8h alo = *(const v8h*)(ap);
        v8h ahi = *(const v8h*)(ap + 8);
        aFrag[s] = __builtin_shufflevector(alo, ahi,
                                           0,1,2,3,4,5,6,7,8,9,10,11,12,13,14,15);
      }
      const int nbase = wc * 64;
#pragma unroll
      for (int t = 0; t < 4; ++t) {
        // B fragment 32x16: lane = column, K contiguous per lane-half
        const _Float16* bp = &lsB[(nbase + t * 16 + l16) * TBK + lh * 16];
        v16h bfrag = *(const v16h*)bp;
#pragma unroll
        for (int s = 0; s < MT; ++s)
          acc[s][t] = __builtin_amdgcn_wmma_f32_16x16x32_f16(
              false, aFrag[s], false, bfrag, (short)0, acc[s][t], false, false);
      }
    }
    __syncthreads();
  }

  // ---- epilogue: BN scale/bias + residual + ReLU ----
  const int ncol0 = nBlock + wc * 64 + l16;
#pragma unroll
  for (int t = 0; t < 4; ++t) {
    const int n = ncol0 + t * 16;
    if (n < Ntot) {
#pragma unroll
      for (int s = 0; s < MT; ++s) {
#pragma unroll
        for (int vv = 0; vv < 8; ++vv) {
          const int m = ocBlock + (wm * MT + s) * 16 + vv + lh * 8;  // M = vgpr + 8*(lane>=16)
          float c = acc[s][t][vv];
          c = c * scale[m] + bias[m];
          if (res) c += res[(size_t)m * Ntot + n];
          if (do_relu) c = fmaxf(c, 0.0f);
          out[(size_t)m * Ntot + n] = c;
        }
      }
    }
  }
}

// ---------------------------------------------------------------------------
// helpers
// ---------------------------------------------------------------------------
__global__ void w2h_kernel(const float* __restrict__ w, _Float16* __restrict__ o, int n) {
  int i = blockIdx.x * blockDim.x + threadIdx.x;
  if (i < n) o[i] = (_Float16)w[i];
}

__global__ void bnfold_kernel(const float* g, const float* b, const float* m,
                              const float* v, float* scale, float* bias, int C) {
  int i = blockIdx.x * blockDim.x + threadIdx.x;
  if (i < C) { float s = g[i] * rsqrtf(v[i] + 1e-5f); scale[i] = s; bias[i] = b[i] - m[i] * s; }
}

// CBHW 3x3 s2 p1 maxpool
__global__ void maxpool_kernel(const float* __restrict__ in, float* __restrict__ out,
                               int C, int B, int H, int W, int OH, int OW) {
  int idx = blockIdx.x * blockDim.x + threadIdx.x;
  int total = C * B * OH * OW;
  if (idx >= total) return;
  int ow = idx % OW; int t = idx / OW;
  int oh = t % OH;  t /= OH;
  int b  = t % B;   int c = t / B;
  const float* p = in + ((size_t)c * B + b) * H * W;
  float m = -INFINITY;
  for (int dh = 0; dh < 3; ++dh) {
    int h = oh * 2 - 1 + dh;
    if (h < 0 || h >= H) continue;
    for (int dw = 0; dw < 3; ++dw) {
      int w = ow * 2 - 1 + dw;
      if (w < 0 || w >= W) continue;
      m = fmaxf(m, p[h * W + w]);
    }
  }
  out[idx] = m;
}

// CBHW [C][B][HW] -> pooled [B][C]
__global__ void avgpool_kernel(const float* __restrict__ in, float* __restrict__ out,
                               int C, int B, int HW) {
  int idx = blockIdx.x * blockDim.x + threadIdx.x;
  if (idx >= B * C) return;
  int c = idx % C, b = idx / C;
  const float* p = in + ((size_t)c * B + b) * HW;
  float s = 0.f;
  for (int i = 0; i < HW; ++i) s += p[i];
  out[(size_t)b * C + c] = s / (float)HW;
}

// y[b][j] = relu?( x[b][:] . w[:][j] + bias[j] )
__global__ void linear_kernel(const float* __restrict__ x, const float* __restrict__ w,
                              const float* __restrict__ bias, float* __restrict__ y,
                              int nB, int K, int J, int do_relu) {
  int idx = blockIdx.x * blockDim.x + threadIdx.x;
  if (idx >= nB * J) return;
  int j = idx % J, b = idx / J;
  float s = bias ? bias[j] : 0.f;
  const float* xp = x + (size_t)b * K;
  for (int k = 0; k < K; ++k) s += xp[k] * w[(size_t)k * J + j];
  if (do_relu) s = fmaxf(s, 0.f);
  y[idx] = s;
}

// one block per (b, patch): 64-bin LDS histogram over 56x56 gray patch
__global__ void hist_kernel(const float* __restrict__ img, float* __restrict__ hist) {
  __shared__ float bins[64];
  const int bp = blockIdx.x;
  const int b = bp >> 4, p = bp & 15;
  const int pr = p >> 2, pc = p & 3;
  if (threadIdx.x < 64) bins[threadIdx.x] = 0.f;
  __syncthreads();
  const int HW = 224 * 224;
  const float* base = img + (size_t)b * 3 * HW;
  for (int i = threadIdx.x; i < 56 * 56; i += blockDim.x) {
    int ii = i / 56, jj = i - ii * 56;
    int off = (pr * 56 + ii) * 224 + (pc * 56 + jj);
    float g = (base[off] + base[HW + off] + base[2 * HW + off]) * (1.0f / 3.0f);
    float v = fminf(fmaxf(g * 255.f, 0.f), 255.f);
    int bi = (int)floorf(v * (64.f / 255.f));
    bi = bi < 0 ? 0 : (bi > 63 ? 63 : bi);
    atomicAdd(&bins[bi], 1.0f);
  }
  __syncthreads();
  if (threadIdx.x < 64)
    hist[(size_t)(b * 16 + p) * 64 + threadIdx.x] = bins[threadIdx.x] / (3136.0f + 1e-6f);
}

// concat [img_feat | hist_feat] per row, L2-normalize (one block per row, 256 thr)
__global__ void featnorm_kernel(const float* __restrict__ img_feat,
                                const float* __restrict__ hist_feat,
                                float* __restrict__ fout) {
  __shared__ float red[256];
  const int b = blockIdx.x, t = threadIdx.x;
  float v = (t < 128) ? img_feat[b * 128 + t] : hist_feat[b * 128 + (t - 128)];
  red[t] = v * v;
  __syncthreads();
  for (int s = 128; s > 0; s >>= 1) { if (t < s) red[t] += red[t + s]; __syncthreads(); }
  float nrm = fmaxf(sqrtf(red[0]), 1e-12f);
  fout[b * 256 + t] = v / nrm;
}

// combined = [f1, f2, |f1-f2|, f1*f2]  -> [B][1024]
__global__ void combine_kernel(const float* __restrict__ f1, const float* __restrict__ f2,
                               float* __restrict__ comb) {
  int idx = blockIdx.x * blockDim.x + threadIdx.x;
  if (idx >= 32 * 1024) return;
  int b = idx >> 10, i = idx & 1023;
  int g = i >> 8, j = i & 255;
  float a = f1[b * 256 + j], c = f2[b * 256 + j];
  float v = (g == 0) ? a : (g == 1) ? c : (g == 2) ? fabsf(a - c) : a * c;
  comb[idx] = v;
}

// ---------------------------------------------------------------------------
static inline void run_conv(hipStream_t s, const float* in, int sb, int sc,
                            int B, int IC, int H, int W, const _Float16* w, int OC,
                            int KH, int KW, int stride, int pad,
                            const float* scale, const float* bias,
                            const float* res, int relu, float* out) {
  const int OH = (H + 2 * pad - KH) / stride + 1;
  const int OW = (W + 2 * pad - KW) / stride + 1;
  const int K = IC * KH * KW;
  const int Ntot = B * OH * OW;
  if (OC % 128 == 0) {
    dim3 grid((Ntot + TBN - 1) / TBN, OC / 128);
    conv_wmma_kernel<2><<<grid, 256, 0, s>>>(in, sb, sc, H, W, w, OC, K, KH * KW, KW,
                                             stride, pad, OH, OW, Ntot,
                                             scale, bias, res, relu, out);
  } else {
    dim3 grid((Ntot + TBN - 1) / TBN, OC / 64);
    conv_wmma_kernel<1><<<grid, 256, 0, s>>>(in, sb, sc, H, W, w, OC, K, KH * KW, KW,
                                             stride, pad, OH, OW, Ntot,
                                             scale, bias, res, relu, out);
  }
}

extern "C" void kernel_launch(void* const* d_in, const int* in_sizes, int n_in,
                              void* d_out, int out_size, void* d_ws, size_t ws_size,
                              hipStream_t stream) {
  (void)in_sizes; (void)n_in; (void)out_size; (void)ws_size;
  const int B = 32;

  // --- input index map: jax pytree (alphabetical dict key) flatten order ---
  // top: img1, img2, params{blocks, c1_b, c1_w, c2_b, c2_w, fc_b, fc_w,
  //                         h1_b, h1_w, h2_b, h2_w, stem_bn, stem_w}
  // block dict: bn1{b,g,m,v}, bn2{b,g,m,v}, conv1, conv2, [down_bn{b,g,m,v}, down_w]
  const int hasDown[8] = {0,0,1,0,1,0,1,0};
  const int outc[8]    = {64,64,128,128,256,256,512,512};
  const int inc[8]     = {64,64,64,128,128,256,256,512};
  const int strides[8] = {1,1,2,1,2,1,2,1};
  int bn1i[8][4], bn2i[8][4], c1i[8], c2i[8], dbni[8][4], dwi[8];
  int idx = 2;
  for (int i = 0; i < 8; ++i) {
    for (int j = 0; j < 4; ++j) bn1i[i][j] = idx++;
    for (int j = 0; j < 4; ++j) bn2i[i][j] = idx++;
    c1i[i] = idx++; c2i[i] = idx++;
    if (hasDown[i]) { for (int j = 0; j < 4; ++j) dbni[i][j] = idx++; dwi[i] = idx++; }
  }
  const int i_c1b = idx++, i_c1w = idx++, i_c2b = idx++, i_c2w = idx++;
  const int i_fcb = idx++, i_fcw = idx++;
  const int i_h1b = idx++, i_h1w = idx++, i_h2b = idx++, i_h2w = idx++;
  int i_sbn[4]; for (int j = 0; j < 4; ++j) i_sbn[j] = idx++;
  const int i_sw = idx++;
  auto F = [&](int i) -> const float* { return (const float*)d_in[i]; };
  const float* img1 = F(0);
  const float* img2 = F(1);

  // --- workspace carve ---
  size_t off = 0;
  auto carve = [&](size_t bytes) -> void* {
    void* p = (char*)d_ws + off; off += (bytes + 255) & ~(size_t)255; return p;
  };
  float* ACT0 = (float*)carve(104ull << 20);            // stem out 64x32x112x112
  float* ACT1 = (float*)carve(26ull << 20);
  float* ACT2 = (float*)carve(26ull << 20);
  float* ACT3 = (float*)carve(26ull << 20);
  _Float16* W16 = (_Float16*)carve(24ull << 20);        // all conv weights as f16
  float* BNBUF = (float*)carve(1ull << 20);             // folded BN scale/bias
  float* pooled    = (float*)carve(B * 512 * 4);
  float* img_feat  = (float*)carve(B * 128 * 4);
  float* hist      = (float*)carve(B * 1024 * 4);
  float* hbuf      = (float*)carve(B * 512 * 4);
  float* hist_feat = (float*)carve(B * 128 * 4);
  float* fnorm1    = (float*)carve(B * 256 * 4);
  float* fnorm2    = (float*)carve(B * 256 * 4);
  float* comb      = (float*)carve(B * 1024 * 4);
  float* c1out     = (float*)carve(B * 256 * 4);

  // --- convert conv weights f32 -> f16 (shared by both branches) ---
  size_t woff = 0;
  auto cvt = [&](int di, int n) -> _Float16* {
    _Float16* p = W16 + woff; woff += (size_t)n;
    w2h_kernel<<<(n + 255) / 256, 256, 0, stream>>>(F(di), p, n);
    return p;
  };
  _Float16* wstem = cvt(i_sw, 64 * 3 * 7 * 7);
  _Float16 *wc1[8], *wc2[8], *wdn[8];
  for (int i = 0; i < 8; ++i) {
    wc1[i] = cvt(c1i[i], outc[i] * inc[i] * 9);
    wc2[i] = cvt(c2i[i], outc[i] * outc[i] * 9);
    wdn[i] = hasDown[i] ? cvt(dwi[i], outc[i] * inc[i]) : nullptr;
  }

  // --- fold BN params: scale = g*rsqrt(v+eps), bias = b - m*scale ---
  size_t boff = 0;
  auto fold = [&](const int bn[4], int C, float** sc, float** bi) {
    *sc = BNBUF + boff; boff += C;
    *bi = BNBUF + boff; boff += C;
    bnfold_kernel<<<(C + 255) / 256, 256, 0, stream>>>(
        F(bn[1]), F(bn[0]), F(bn[2]), F(bn[3]), *sc, *bi, C);
  };
  float *stem_sc, *stem_bi;
  fold(i_sbn, 64, &stem_sc, &stem_bi);
  float *sc1[8], *bi1[8], *sc2[8], *bi2[8], *scd[8], *bid[8];
  for (int i = 0; i < 8; ++i) {
    fold(bn1i[i], outc[i], &sc1[i], &bi1[i]);
    fold(bn2i[i], outc[i], &sc2[i], &bi2[i]);
    if (hasDown[i]) fold(dbni[i], outc[i], &scd[i], &bid[i]);
  }

  // --- one full siamese branch ---
  auto branch = [&](const float* img, float* fnorm_out) {
    // stem: NCHW input (sb = 3*224*224, sc = 224*224) -> ACT0 (CBHW 64x32x112x112)
    run_conv(stream, img, 3 * 224 * 224, 224 * 224, B, 3, 224, 224,
             wstem, 64, 7, 7, 2, 3, stem_sc, stem_bi, nullptr, 1, ACT0);
    // maxpool 3x3 s2 p1 -> ACT1 (64x32x56x56)
    {
      int total = 64 * B * 56 * 56;
      maxpool_kernel<<<(total + 255) / 256, 256, 0, stream>>>(ACT0, ACT1, 64, B, 112, 112, 56, 56);
    }
    float *p1 = ACT1, *p2 = ACT2, *p3 = ACT3;
    int C = 64, H = 56, W = 56;
    for (int i = 0; i < 8; ++i) {
      const int OC = outc[i], st = strides[i];
      const int OH = (H + 2 - 3) / st + 1, OW = (W + 2 - 3) / st + 1;
      // conv1 + bn1 + relu   (CBHW input: sb = H*W, sc = B*H*W)
      run_conv(stream, p1, H * W, B * H * W, B, C, H, W,
               wc1[i], OC, 3, 3, st, 1, sc1[i], bi1[i], nullptr, 1, p2);
      const float* res; float* ob;
      if (hasDown[i]) {
        run_conv(stream, p1, H * W, B * H * W, B, C, H, W,
                 wdn[i], OC, 1, 1, st, 0, scd[i], bid[i], nullptr, 0, p3);
        res = p3; ob = p1;                 // x no longer needed -> overwrite
      } else {
        res = p1; ob = p3;
      }
      // conv2 + bn2 + residual + relu
      run_conv(stream, p2, OH * OW, B * OH * OW, B, OC, OH, OW,
               wc2[i], OC, 3, 3, 1, 1, sc2[i], bi2[i], res, 1, ob);
      if (!hasDown[i]) { float* t = p1; p1 = p3; p3 = t; }   // new x = ob
      C = OC; H = OH; W = OW;
    }
    // global avg pool [512][B][49] -> pooled [B][512]
    avgpool_kernel<<<(B * 512 + 255) / 256, 256, 0, stream>>>(p1, pooled, 512, B, 49);
    // fc: [B,512] @ [512,128] + b
    linear_kernel<<<(B * 128 + 255) / 256, 256, 0, stream>>>(
        pooled, F(i_fcw), F(i_fcb), img_feat, B, 512, 128, 0);
    // histogram branch
    hist_kernel<<<B * 16, 256, 0, stream>>>(img, hist);
    linear_kernel<<<(B * 512 + 255) / 256, 256, 0, stream>>>(
        hist, F(i_h1w), F(i_h1b), hbuf, B, 1024, 512, 1);
    linear_kernel<<<(B * 128 + 255) / 256, 256, 0, stream>>>(
        hbuf, F(i_h2w), F(i_h2b), hist_feat, B, 512, 128, 0);
    // concat + L2 normalize -> [B,256]
    featnorm_kernel<<<B, 256, 0, stream>>>(img_feat, hist_feat, fnorm_out);
  };

  branch(img1, fnorm1);
  branch(img2, fnorm2);

  // head
  combine_kernel<<<(B * 1024 + 255) / 256, 256, 0, stream>>>(fnorm1, fnorm2, comb);
  linear_kernel<<<(B * 256 + 255) / 256, 256, 0, stream>>>(
      comb, F(i_c1w), F(i_c1b), c1out, B, 1024, 256, 1);
  linear_kernel<<<(B * 1 + 255) / 256, 256, 0, stream>>>(
      c1out, F(i_c2w), F(i_c2b), (float*)d_out, B, 256, 1, 0);
}
